// Allignement_49340584296722
// MI455X (gfx1250) — matrix operand order repaired and checked by reading, not compiled, not run on hardware
//
#include <hip/hip_runtime.h>
#include <hip/hip_bf16.h>
#include <math.h>

// ---------------------------------------------------------------------------
// Bahdanau additive attention, MI455X (gfx1250, wave32, WMMA).
//   B=1, S=65536, H=512, A=256.
// Pass 1 (dominant, 17.2 GFLOP): si@Wa.T via v_wmma_f32_16x16x32_bf16 with
//   fused tanh/va reduction (~memory floor: one 128MB si stream).
// Pass 2: softmax (deterministic fixed-order reductions).
// Pass 3: context = alpha@si, second 128MB si stream done with CDNA5
//   GLOBAL_LOAD_ASYNC_TO_LDS_B128 double buffering (ASYNCcnt pipeline).
// Workspace use: ~1.1 MB of d_ws.
// ---------------------------------------------------------------------------

typedef __attribute__((ext_vector_type(16))) __bf16 v16bf;
typedef __attribute__((ext_vector_type(8)))  __bf16 v8bf;
typedef __attribute__((ext_vector_type(8)))  float  v8f;
typedef __attribute__((ext_vector_type(4)))  float  f4;

#define S_LEN 65536
#define HID   512
#define ADIM  256
#define ROWS  32                          // si rows per score block
#define SCORE_BLOCKS (S_LEN / ROWS)       // 2048
#define EXP_BLOCKS   64
#define EXP_PER_T    4                    // 64*256*4 = 65536
#define CTX_BLOCKS   128
#define CTX_ROWS     (S_LEN / CTX_BLOCKS) // 512
#define CCHUNK       8                    // rows per async chunk (16 KB)
#define NCHUNK       (CTX_ROWS / CCHUNK)  // 64

// workspace layout (float offsets)
#define WS_ENCP   0        // 256   : enc_proj
#define WS_GMAX   256      // 1     : global max of e
#define WS_GSUM   257      // 1     : global sum of exp
#define WS_E      512      // 65536 : raw scores e[s]
#define WS_AUN    66048    // 65536 : unnormalized exp(e - max)
#define WS_MAXP   131584   // 2048  : per-block maxima
#define WS_SUMP   133632   // 64    : per-block exp sums
#define WS_CTXP   133696   // 128*512 : partial contexts (ends 199232)
#define WS_WABF   200704   // bf16 Wa as 131072 ushort (ends ~266240 floats)

__device__ __forceinline__ unsigned short f2bf(float f) {
    union { float f; unsigned u; } x; x.f = f;
    unsigned r = x.u + 0x7FFFu + ((x.u >> 16) & 1u);   // round-to-nearest-even
    return (unsigned short)(r >> 16);
}

// ---------------------------------------------------------------------------
// K0: enc_proj[a] = sum_h enc_out[h] * Ua[a,h]     (1 block, 256 threads)
// ---------------------------------------------------------------------------
__global__ void encproj_kernel(const float* __restrict__ enc,
                               const float* __restrict__ Ua,
                               float* __restrict__ encp) {
    int a = threadIdx.x;                       // 0..255
    const f4* u  = (const f4*)(Ua + (size_t)a * (2 * HID));
    const f4* e4 = (const f4*)enc;
    float s = 0.f;
    for (int i = 0; i < (2 * HID) / 4; ++i) {
        f4 uu = u[i], ee = e4[i];
        s += uu.x * ee.x + uu.y * ee.y + uu.z * ee.z + uu.w * ee.w;
    }
    encp[a] = s;
}

// ---------------------------------------------------------------------------
// K1: Wa fp32 -> bf16 copy (stays resident in L2; 256 KB)
// ---------------------------------------------------------------------------
__global__ void cvtwa_kernel(const float* __restrict__ Wa,
                             unsigned short* __restrict__ wab) {
    int i = blockIdx.x * 256 + threadIdx.x;    // grid 512 -> 131072 elems
    wab[i] = f2bf(Wa[i]);
}

// ---------------------------------------------------------------------------
// K2: score kernel.  e[s] = sum_a tanh(si[s,:]@Wa[a,:] + encp[a]) * va[a]
// 512 threads = 16 waves; wave w owns A-column tile w; 2 row tiles of 16.
// ---------------------------------------------------------------------------
__global__ void __launch_bounds__(512)
score_kernel(const float* __restrict__ si,
             const unsigned short* __restrict__ wab,   // bf16 Wa [256][512]
             const float* __restrict__ encp,
             const float* __restrict__ va,
             float* __restrict__ escore,
             float* __restrict__ maxpart) {
    __shared__ __align__(32) unsigned short si_lds[ROWS][HID];  // 32 KB bf16
    __shared__ float rowsum[16][ROWS];                          // 2 KB

    const int t    = threadIdx.x;
    const int lane = t & 31;
    const int wv   = t >> 5;          // wave id = A column tile (0..15)
    const int n    = lane & 15;       // N within tile / row-M within A tile
    const int g    = lane >> 4;       // lane half

    // ---- stage si row tile, converting fp32 -> bf16 (coalesced float4) ----
    const f4* s4 = (const f4*)(si + (size_t)blockIdx.x * ROWS * HID);
    for (int v = t; v < ROWS * (HID / 4); v += 512) {
        f4 x = s4[v];
        int row = v >> 7;             // HID/4 = 128 float4 per row
        int c   = (v & 127) << 2;
        si_lds[row][c + 0] = f2bf(x.x);
        si_lds[row][c + 1] = f2bf(x.y);
        si_lds[row][c + 2] = f2bf(x.z);
        si_lds[row][c + 3] = f2bf(x.w);
    }
    __syncthreads();

    const int col = wv * 16 + n;          // attn dim owned by this lane
    const float ep = encp[col];
    const float vv = va[col];

    v8f acc0 = {0.f,0.f,0.f,0.f,0.f,0.f,0.f,0.f};
    v8f acc1 = {0.f,0.f,0.f,0.f,0.f,0.f,0.f,0.f};

    const unsigned short* wrow = wab + (size_t)col * HID;

    for (int ks = 0; ks < HID / 32; ++ks) {
        const int k0 = ks * 32;
        // B-frag (Wa^T): 16 contiguous bf16 from row `col`, half-selected by g
        v16bf bfrag = *(const v16bf*)(const void*)(wrow + k0 + 16 * g);
        // A-frags: contiguous K order matching B (element s -> K=16g+s)
        union { v16bf v; v8bf h[2]; } a0, a1;
        a0.h[0] = *(const v8bf*)(const void*)&si_lds[n     ][k0 +     16 * g];
        a0.h[1] = *(const v8bf*)(const void*)&si_lds[n     ][k0 + 8 + 16 * g];
        a1.h[0] = *(const v8bf*)(const void*)&si_lds[16 + n][k0 +     16 * g];
        a1.h[1] = *(const v8bf*)(const void*)&si_lds[16 + n][k0 + 8 + 16 * g];
        acc0 = __builtin_amdgcn_wmma_f32_16x16x32_bf16(false, a0.v, false, bfrag,
                                                       (short)0, acc0, false, false);
        acc1 = __builtin_amdgcn_wmma_f32_16x16x32_bf16(false, a1.v, false, bfrag,
                                                       (short)0, acc1, false, false);
    }

    // ---- fused tanh(x+enc)*va, reduce over N (lanes in same half) ----------
    // C/D layout: VGPR i, lane half g -> M = i + 8g ; N = lane%16.
    #pragma unroll
    for (int rt = 0; rt < 2; ++rt) {
        v8f acc = rt ? acc1 : acc0;
        #pragma unroll
        for (int i = 0; i < 8; ++i) {
            float p = tanhf(acc[i] + ep) * vv;
            p += __shfl_xor(p, 1, 32);
            p += __shfl_xor(p, 2, 32);
            p += __shfl_xor(p, 4, 32);
            p += __shfl_xor(p, 8, 32);
            if (n == 0) rowsum[wv][rt * 16 + i + 8 * g] = p;
        }
    }
    __syncthreads();

    // ---- cross-wave reduce (fixed order -> deterministic), block max -------
    if (t < ROWS) {
        float e = 0.f;
        #pragma unroll
        for (int w = 0; w < 16; ++w) e += rowsum[w][t];
        escore[blockIdx.x * ROWS + t] = e;
        float m = e;
        m = fmaxf(m, __shfl_xor(m, 1,  32));
        m = fmaxf(m, __shfl_xor(m, 2,  32));
        m = fmaxf(m, __shfl_xor(m, 4,  32));
        m = fmaxf(m, __shfl_xor(m, 8,  32));
        m = fmaxf(m, __shfl_xor(m, 16, 32));
        if (t == 0) maxpart[blockIdx.x] = m;
    }
}

// ---------------------------------------------------------------------------
// K3: global max of 2048 partials (1 block)
// ---------------------------------------------------------------------------
__global__ void reduce_max_kernel(const float* __restrict__ maxp,
                                  float* __restrict__ gmax) {
    __shared__ float sm[8];
    float m = -3.4e38f;
    for (int i = threadIdx.x; i < SCORE_BLOCKS; i += 256) m = fmaxf(m, maxp[i]);
    m = fmaxf(m, __shfl_xor(m, 1,  32));
    m = fmaxf(m, __shfl_xor(m, 2,  32));
    m = fmaxf(m, __shfl_xor(m, 4,  32));
    m = fmaxf(m, __shfl_xor(m, 8,  32));
    m = fmaxf(m, __shfl_xor(m, 16, 32));
    if ((threadIdx.x & 31) == 0) sm[threadIdx.x >> 5] = m;
    __syncthreads();
    if (threadIdx.x == 0) {
        float r = sm[0];
        for (int w = 1; w < 8; ++w) r = fmaxf(r, sm[w]);
        *gmax = r;
    }
}

// ---------------------------------------------------------------------------
// K4: exp(e - gmax), per-block partial sums
// ---------------------------------------------------------------------------
__global__ void expsum_kernel(const float* __restrict__ e,
                              const float* __restrict__ gmaxp,
                              float* __restrict__ aun,
                              float* __restrict__ sump) {
    __shared__ float sm[8];
    const int t = threadIdx.x;
    const int base = blockIdx.x * 256 * EXP_PER_T;
    const float gmax = *gmaxp;
    float s = 0.f;
    #pragma unroll
    for (int j = 0; j < EXP_PER_T; ++j) {
        int idx = base + j * 256 + t;
        float p = __expf(e[idx] - gmax);
        aun[idx] = p;
        s += p;
    }
    s += __shfl_xor(s, 1,  32);
    s += __shfl_xor(s, 2,  32);
    s += __shfl_xor(s, 4,  32);
    s += __shfl_xor(s, 8,  32);
    s += __shfl_xor(s, 16, 32);
    if ((t & 31) == 0) sm[t >> 5] = s;
    __syncthreads();
    if (t == 0) {
        float r = 0.f;
        for (int w = 0; w < 8; ++w) r += sm[w];
        sump[blockIdx.x] = r;
    }
}

// ---------------------------------------------------------------------------
// K5: sum of 64 partials (1 wave)
// ---------------------------------------------------------------------------
__global__ void reduce_sum_kernel(const float* __restrict__ sump,
                                  float* __restrict__ gsum) {
    const int t = threadIdx.x;   // 32 threads
    float s = sump[t] + sump[t + 32];
    s += __shfl_xor(s, 1,  32);
    s += __shfl_xor(s, 2,  32);
    s += __shfl_xor(s, 4,  32);
    s += __shfl_xor(s, 8,  32);
    s += __shfl_xor(s, 16, 32);
    if (t == 0) *gsum = s;
}

// ---------------------------------------------------------------------------
// Async-copy helpers (CDNA5 GLOBAL_LOAD_ASYNC_TO_LDS, tracked by ASYNCcnt)
// ---------------------------------------------------------------------------
__device__ __forceinline__ void async_copy_b128(unsigned lds_byte_addr,
                                                const float* gptr) {
    unsigned long long ga = (unsigned long long)(size_t)gptr;
    asm volatile("global_load_async_to_lds_b128 %0, %1, off"
                 :: "v"(lds_byte_addr), "v"(ga) : "memory");
}

__device__ __forceinline__ void wait_asynccnt_le4() {
#if defined(__has_builtin) && __has_builtin(__builtin_amdgcn_s_wait_asynccnt)
    __builtin_amdgcn_s_wait_asynccnt(4);
#else
    asm volatile("s_wait_asynccnt 4" ::: "memory");
#endif
}
__device__ __forceinline__ void wait_asynccnt_le0() {
#if defined(__has_builtin) && __has_builtin(__builtin_amdgcn_s_wait_asynccnt)
    __builtin_amdgcn_s_wait_asynccnt(0);
#else
    asm volatile("s_wait_asynccnt 0" ::: "memory");
#endif
}

// ---------------------------------------------------------------------------
// K6: normalized alpha out + partial context. Second 128MB si stream done as
// an async double-buffered LDS pipeline: while chunk c is accumulated from
// LDS, chunk c+1 streams in via GLOBAL_LOAD_ASYNC_TO_LDS_B128 (ASYNCcnt).
// ---------------------------------------------------------------------------
__global__ void __launch_bounds__(256)
ctx_partial_kernel(const float* __restrict__ si,
                   const float* __restrict__ aun,
                   const float* __restrict__ gsum,
                   float* __restrict__ alpha_out,
                   float* __restrict__ ctxp) {
    __shared__ __align__(16) float sbuf[2][CCHUNK][HID];  // 2 x 16 KB
    __shared__ float al[CTX_ROWS];                        // 2 KB

    const int b = blockIdx.x, t = threadIdx.x;
    const int base = b * CTX_ROWS;
    const float inv = 1.f / *gsum;
    for (int r = t; r < CTX_ROWS; r += 256) {
        float a = aun[base + r];
        al[r] = a;
        alpha_out[base + r] = a * inv;
    }

    const float* gbase = si + (size_t)base * HID;

    // stage one chunk: CCHUNK*HID floats, 4 x b128 per thread
    auto stage = [&](int ch, int buf) {
        #pragma unroll
        for (int i = 0; i < 4; ++i) {
            int v   = t + i * 256;          // float4 index within chunk
            int row = v >> 7;               // HID/4 = 128 float4 per row
            int c4  = v & 127;
            unsigned lds = (unsigned)(size_t)&sbuf[buf][row][c4 * 4];
            async_copy_b128(lds, gbase + ((size_t)ch * CCHUNK * HID) + (size_t)v * 4);
        }
    };

    stage(0, 0);
    float c0 = 0.f, c1 = 0.f;
    for (int c = 0; c < NCHUNK; ++c) {
        const int buf = c & 1;
        if (c + 1 < NCHUNK) {
            stage(c + 1, buf ^ 1);
            wait_asynccnt_le4();            // chunk c's 4 loads retired (in order)
        } else {
            wait_asynccnt_le0();
        }
        __syncthreads();                    // chunk c visible block-wide
        #pragma unroll
        for (int r = 0; r < CCHUNK; ++r) {
            float a = al[c * CCHUNK + r];
            c0 += a * sbuf[buf][r][t];
            c1 += a * sbuf[buf][r][t + 256];
        }
        __syncthreads();                    // done reading buf before restage
    }
    ctxp[b * HID + t]       = c0;
    ctxp[b * HID + t + 256] = c1;
}

// ---------------------------------------------------------------------------
// K7: reduce partial contexts, scale by 1/gsum -> context (1 block, 512 thr)
// ---------------------------------------------------------------------------
__global__ void ctx_reduce_kernel(const float* __restrict__ ctxp,
                                  const float* __restrict__ gsum,
                                  float* __restrict__ out) {
    const int h = threadIdx.x;               // 0..511
    float s = 0.f;
    for (int b = 0; b < CTX_BLOCKS; ++b) s += ctxp[b * HID + h];
    out[h] = s * (1.f / *gsum);
}

// ---------------------------------------------------------------------------
extern "C" void kernel_launch(void* const* d_in, const int* in_sizes, int n_in,
                              void* d_out, int out_size, void* d_ws, size_t ws_size,
                              hipStream_t stream) {
    const float* enc = (const float*)d_in[0];   // [1, 1024]
    const float* si  = (const float*)d_in[1];   // [1, 65536, 512]
    const float* Wa  = (const float*)d_in[2];   // [256, 512]
    const float* Ua  = (const float*)d_in[3];   // [256, 1024]
    const float* va  = (const float*)d_in[4];   // [256]
    float* out = (float*)d_out;                 // [512 context | 65536 alpha]
    float* ws  = (float*)d_ws;
    unsigned short* wab = (unsigned short*)(ws + WS_WABF);

    encproj_kernel<<<1, 256, 0, stream>>>(enc, Ua, ws + WS_ENCP);
    cvtwa_kernel<<<(ADIM * HID) / 256, 256, 0, stream>>>(Wa, wab);
    score_kernel<<<SCORE_BLOCKS, 512, 0, stream>>>(si, wab, ws + WS_ENCP, va,
                                                   ws + WS_E, ws + WS_MAXP);
    reduce_max_kernel<<<1, 256, 0, stream>>>(ws + WS_MAXP, ws + WS_GMAX);
    expsum_kernel<<<EXP_BLOCKS, 256, 0, stream>>>(ws + WS_E, ws + WS_GMAX,
                                                  ws + WS_AUN, ws + WS_SUMP);
    reduce_sum_kernel<<<1, 32, 0, stream>>>(ws + WS_SUMP, ws + WS_GSUM);
    ctx_partial_kernel<<<CTX_BLOCKS, 256, 0, stream>>>(si, ws + WS_AUN,
                                                       ws + WS_GSUM,
                                                       out + HID, ws + WS_CTXP);
    ctx_reduce_kernel<<<1, 512, 0, stream>>>(ws + WS_CTXP, ws + WS_GSUM, out);
}
